// LSTMCell_X_34462817583251
// MI455X (gfx1250) — matrix operand order repaired and checked by reading, not compiled, not run on hardware
//
#include <hip/hip_runtime.h>

typedef __bf16 bf16;
typedef __bf16 bf16x8  __attribute__((ext_vector_type(8)));
typedef __bf16 bf16x16 __attribute__((ext_vector_type(16)));
typedef float  f32x8   __attribute__((ext_vector_type(8)));

#define B_SZ 32768
#define I_SZ 512
#define O_SZ 512
#define K_SZ 1024   // I + O
#define G_SZ 2048   // 4 * O

__device__ __forceinline__ bf16x16 cat8(bf16x8 lo, bf16x8 hi) {
  return __builtin_shufflevector(lo, hi, 0,1,2,3,4,5,6,7,8,9,10,11,12,13,14,15);
}

// ---------------- pre-pass: f32 -> bf16 ----------------
// Xb[b][k] = concat(input, h)[b][k] in bf16, contiguous K.
__global__ __launch_bounds__(256) void k_cvt_x(const float* __restrict__ xin,
                                               const float* __restrict__ hin,
                                               bf16* __restrict__ Xb) {
  size_t t   = (size_t)blockIdx.x * blockDim.x + threadIdx.x;
  size_t idx = t * 8;                 // element index in [B, 1024]
  size_t b   = idx >> 10;
  int    i   = (int)(idx & 1023);
  const float* src = (i < I_SZ) ? (xin + b * I_SZ + i)
                                : (hin + b * O_SZ + (i - I_SZ));
  float4 v0 = *(const float4*)(src);
  float4 v1 = *(const float4*)(src + 4);
  bf16x8 o;
  o[0] = (bf16)v0.x; o[1] = (bf16)v0.y; o[2] = (bf16)v0.z; o[3] = (bf16)v0.w;
  o[4] = (bf16)v1.x; o[5] = (bf16)v1.y; o[6] = (bf16)v1.z; o[7] = (bf16)v1.w;
  *(bf16x8*)(Xb + idx) = o;
}

__global__ __launch_bounds__(256) void k_cvt_w(const float* __restrict__ W,
                                               bf16* __restrict__ Wb) {
  size_t t   = (size_t)blockIdx.x * blockDim.x + threadIdx.x;
  size_t idx = t * 8;
  float4 v0 = *(const float4*)(W + idx);
  float4 v1 = *(const float4*)(W + idx + 4);
  bf16x8 o;
  o[0] = (bf16)v0.x; o[1] = (bf16)v0.y; o[2] = (bf16)v0.z; o[3] = (bf16)v0.w;
  o[4] = (bf16)v1.x; o[5] = (bf16)v1.y; o[6] = (bf16)v1.z; o[7] = (bf16)v1.w;
  *(bf16x8*)(Wb + idx) = o;
}

// ---------------- fused GEMM + LayerNorm + gates ----------------
// Grid: B/32 workgroups of 512 threads (16 waves).
// Workgroup: rows [32*blk, 32*blk+32), all 2048 gate columns.
// Wave wv: output cols [32*wv, 32*wv+32), taken from all 4 gate blocks:
//   col-tile (gate, j): g = gate*512 + wv*32 + j*16   (gate=0..3, j=0..1)
// => f,i,o,h gates for an output element all live in the same lane.
__global__ __launch_bounds__(512, 1) void k_lstm(
    const bf16* __restrict__ Xb, const bf16* __restrict__ Wb,
    const float* __restrict__ c, const float* __restrict__ gamma,
    const float* __restrict__ beta, float* __restrict__ out) {
  const int tid  = threadIdx.x;
  const int lane = tid & 31;
  const int wv   = tid >> 5;      // wave 0..15
  const int l16  = lane & 15;
  const int hi   = lane >> 4;     // half-wave: 0 or 1
  const int m0   = blockIdx.x * 32;

  const int koffA = hi * 8;       // 16-bit A 16x32 layout (ISA 7.12.2)
  const int koffB = hi * 8;       // 16-bit B 32x16 layout (same k order as A)

  // A row pointers: lane l16 holds row M=l16 of each 16-row tile.
  const bf16* __restrict__ xr0 = Xb + (size_t)(m0 + l16) * K_SZ + koffA;        // rt=0
  const bf16* __restrict__ xr1 = xr0 + (size_t)16 * K_SZ;                        // rt=1
  // B base: lane l16 holds column N=l16 (g row of W); gate/j folded into imm offsets.
  const bf16* __restrict__ wb0 = Wb + (size_t)(wv * 32 + l16) * K_SZ + koffB;

  f32x8 acc[2][4][2] = {};

  for (int kk = 0; kk < K_SZ; kk += 32) {
    bf16x16 a0 = cat8(*(const bf16x8*)(xr0 + kk), *(const bf16x8*)(xr0 + kk + 16));
    bf16x16 a1 = cat8(*(const bf16x8*)(xr1 + kk), *(const bf16x8*)(xr1 + kk + 16));
    #pragma unroll
    for (int g = 0; g < 4; ++g) {
      #pragma unroll
      for (int j = 0; j < 2; ++j) {
        const bf16* p = wb0 + (size_t)(g * O_SZ + j * 16) * K_SZ + kk;
        bf16x16 bfr = cat8(*(const bf16x8*)(p), *(const bf16x8*)(p + 16));
        acc[0][g][j] = __builtin_amdgcn_wmma_f32_16x16x32_bf16(
            false, a0, false, bfr, (short)0, acc[0][g][j], false, false);
        acc[1][g][j] = __builtin_amdgcn_wmma_f32_16x16x32_bf16(
            false, a1, false, bfr, (short)0, acc[1][g][j], false, false);
      }
    }
  }

  // ---- LayerNorm reduction over all 2048 cols per row ----
  // C/D layout: lane (hi,l16), VGPR r  ->  row m = 16*rt + 8*hi + r, col = ... + l16
  __shared__ float s_sum[32];
  __shared__ float s_sq[32];
  if (tid < 32) { s_sum[tid] = 0.0f; s_sq[tid] = 0.0f; }
  __syncthreads();

  #pragma unroll
  for (int rt = 0; rt < 2; ++rt) {
    #pragma unroll
    for (int r = 0; r < 8; ++r) {
      float s = 0.0f, q = 0.0f;
      #pragma unroll
      for (int g = 0; g < 4; ++g)
        #pragma unroll
        for (int j = 0; j < 2; ++j) {
          float v = acc[rt][g][j][r];
          s += v; q += v * v;
        }
      // butterfly across the 16 lanes sharing this row
      #pragma unroll
      for (int msk = 1; msk < 16; msk <<= 1) {
        s += __shfl_xor(s, msk, 32);
        q += __shfl_xor(q, msk, 32);
      }
      if (l16 == 0) {
        int m = rt * 16 + hi * 8 + r;
        atomicAdd(&s_sum[m], s);
        atomicAdd(&s_sq[m], q);
      }
    }
  }
  __syncthreads();

  // ---- normalize + gates, all in-lane ----
  float gmv[4][2], btv[4][2];
  #pragma unroll
  for (int g = 0; g < 4; ++g)
    #pragma unroll
    for (int j = 0; j < 2; ++j) {
      int col = g * O_SZ + wv * 32 + j * 16 + l16;
      gmv[g][j] = gamma[col];
      btv[g][j] = beta[col];
    }

  const float inv = 1.0f / (float)G_SZ;
  #pragma unroll
  for (int rt = 0; rt < 2; ++rt) {
    #pragma unroll
    for (int r = 0; r < 8; ++r) {
      int m = rt * 16 + hi * 8 + r;
      float mean = s_sum[m] * inv;
      float var  = s_sq[m] * inv - mean * mean;
      float rstd = rsqrtf(var + 1e-5f);
      size_t row = (size_t)(m0 + m);
      #pragma unroll
      for (int j = 0; j < 2; ++j) {
        float zf = (acc[rt][0][j][r] - mean) * rstd * gmv[0][j] + btv[0][j];
        float zi = (acc[rt][1][j][r] - mean) * rstd * gmv[1][j] + btv[1][j];
        float zo = (acc[rt][2][j][r] - mean) * rstd * gmv[2][j] + btv[2][j];
        float zh = (acc[rt][3][j][r] - mean) * rstd * gmv[3][j] + btv[3][j];
        float fg = 1.0f / (1.0f + __expf(-zf));
        float ig = 1.0f / (1.0f + __expf(-zi));
        float og = 1.0f / (1.0f + __expf(-zo));
        float hg = 0.5f * zh * (1.0f + erff(zh * 0.70710678118654752f)); // exact GELU
        size_t cidx = row * O_SZ + (size_t)(wv * 32 + j * 16 + l16);
        float cell = fg * c[cidx] + ig * hg;
        out[cidx] = og * cell;                        // output
        out[cidx + (size_t)B_SZ * O_SZ] = cell;       // cell state
      }
    }
  }
}

extern "C" void kernel_launch(void* const* d_in, const int* in_sizes, int n_in,
                              void* d_out, int out_size, void* d_ws, size_t ws_size,
                              hipStream_t stream) {
  const float* inp   = (const float*)d_in[0];
  const float* h     = (const float*)d_in[1];
  const float* c     = (const float*)d_in[2];
  const float* W     = (const float*)d_in[3];
  const float* gamma = (const float*)d_in[4];
  const float* beta  = (const float*)d_in[5];
  float* out = (float*)d_out;

  bf16* Xb = (bf16*)d_ws;                       // [B, 1024] bf16  (64 MB)
  bf16* Wb = Xb + (size_t)B_SZ * K_SZ;          // [2048, 1024] bf16 (4 MB)

  k_cvt_x<<<(B_SZ * K_SZ / 8) / 256, 256, 0, stream>>>(inp, h, Xb);   // 16384 blocks
  k_cvt_w<<<(G_SZ * K_SZ / 8) / 256, 256, 0, stream>>>(W, Wb);        // 1024 blocks
  k_lstm<<<B_SZ / 32, 512, 0, stream>>>(Xb, Wb, c, gamma, beta, out); // 1024 blocks
}
